// ClsTransformer_1013612282427
// MI455X (gfx1250) — matrix operand order repaired and checked by reading, not compiled, not run on hardware
//
#include <hip/hip_runtime.h>
#include <stdint.h>

typedef __attribute__((ext_vector_type(8)))  float  v8f;
typedef __bf16 bf16;
typedef __attribute__((ext_vector_type(8)))  __bf16 v8bf;
typedef __attribute__((ext_vector_type(16))) __bf16 v16bf;
typedef __attribute__((ext_vector_type(4)))  unsigned int v4u;
typedef __attribute__((ext_vector_type(8)))  int v8i;
typedef __attribute__((ext_vector_type(4)))  int v4i;

constexpr int CB = 8, CN = 1024, CD = 128, CH = 8, CMID = 2048, CL = 3, CK = 30;
// per-layer bf16 weight block layout (elements):
constexpr size_t W_QT = 0, W_KT = 65536, W_VT = 131072, W_OT = 196608, W_F1 = 262144, W_F2 = 524288;
constexpr size_t W_LAYER = 786432;

#if defined(__HIP_DEVICE_COMPILE__) && __has_builtin(__builtin_amdgcn_tensor_load_to_lds) && __has_builtin(__builtin_amdgcn_s_wait_tensorcnt)
#define USE_TDM 1
#else
#define USE_TDM 0
#endif

static __device__ __forceinline__ v8f wmma_bf16(v16bf a, v16bf b, v8f c) {
  return __builtin_amdgcn_wmma_f32_16x16x32_bf16(false, a, false, b, (short)0, c, false, false);
}

// A-operand (16xK tile, row-major source): lane = row (0..15), half-lane p:
// contiguous K-slice p*16 .. p*16+15
static __device__ __forceinline__ v16bf loadA16(const bf16* base, int stride, int lane, int kk) {
  int row = lane & 15, p = lane >> 4;
  const bf16* ptr = base + (size_t)row * stride + kk + p * 16;
  v8bf c0 = *(const v8bf*)ptr;
  v8bf c1 = *(const v8bf*)(ptr + 8);
  v16bf r;
#pragma unroll
  for (int i = 0; i < 8; i++) { r[i] = c0[i]; r[i + 8] = c1[i]; }
  return r;
}
// B-operand (Kx16 tile, source stored transposed [16 rows][K]): lane = column,
// half-lane p: contiguous K-slice p*16 .. p*16+15
static __device__ __forceinline__ v16bf loadB16(const bf16* base, int stride, int lane, int kk) {
  int row = lane & 15, p = lane >> 4;
  const bf16* ptr = base + (size_t)row * stride + kk + p * 16;
  v8bf c0 = *(const v8bf*)ptr;
  v8bf c1 = *(const v8bf*)(ptr + 8);
  v16bf r;
#pragma unroll
  for (int i = 0; i < 8; i++) { r[i] = c0[i]; r[i + 8] = c1[i]; }
  return r;
}
static __device__ __forceinline__ float wsum(float v) {
#pragma unroll
  for (int m = 16; m; m >>= 1) v += __shfl_xor(v, m, 32);
  return v;
}

#if USE_TDM
// Issue a TDM 2-D tile load: global (row-major, 2-byte elems, row stride
// `stride_el`) -> LDS (packed row-major tile_d1 x tile_d0). D# per ISA 8.3/8.4.
static __device__ __forceinline__ void tdm_load_2d(unsigned int lds_off, const void* gptr,
                                                   unsigned int tile_d0, unsigned int tile_d1,
                                                   unsigned int tensor_d0, unsigned int tensor_d1,
                                                   unsigned long long stride_el) {
  unsigned long long ga = (unsigned long long)(uintptr_t)gptr;
  v4u g0;
  g0[0] = 1u;                                            // count=1, user mode
  g0[1] = lds_off;                                       // lds_addr
  g0[2] = (unsigned int)(ga & 0xFFFFFFFFu);              // global_addr[31:0]
  g0[3] = (unsigned int)((ga >> 32) & 0x1FFFFFFu) | 0x80000000u;  // addr[56:32] | type=2
  v8i g1;
  g1[0] = 0x00010000;                                    // data_size = 1 (2 bytes)
  g1[1] = (int)((tensor_d0 & 0xFFFFu) << 16);            // tensor_dim0[15:0]
  g1[2] = (int)(((tensor_d0 >> 16) & 0xFFFFu) | ((tensor_d1 & 0xFFFFu) << 16));
  g1[3] = (int)(((tensor_d1 >> 16) & 0xFFFFu) | ((tile_d0 & 0xFFFFu) << 16));
  g1[4] = (int)(tile_d1 & 0xFFFFu);                      // tile_dim1, tile_dim2=0
  g1[5] = (int)(unsigned int)(stride_el & 0xFFFFFFFFu);  // tensor_dim0_stride[31:0]
  g1[6] = (int)(unsigned int)((stride_el >> 32) & 0xFFFFu);
  g1[7] = 0;
  v4i z4 = {0, 0, 0, 0};
  v8i z8 = {0, 0, 0, 0, 0, 0, 0, 0};
  __builtin_amdgcn_tensor_load_to_lds(g0, g1, z4, z4, z8, 0);
}
static __device__ __forceinline__ unsigned int lds_off_of(const void* p) {
  return (unsigned int)(uintptr_t)p;  // flat LDS aperture: addr[31:0] == LDS offset
}
#endif

// ---------------- weight convert + transpose: out[N][K] = (bf16) in[K][N] ----------------
__global__ void k_transcvt(const float* __restrict__ in, bf16* __restrict__ out, int K, int Nout) {
  int idx = blockIdx.x * 256 + threadIdx.x;
  if (idx >= K * Nout) return;
  int n = idx / K, k = idx - n * K;
  out[idx] = (bf16)in[(size_t)k * Nout + n];
}

// ---------------- init: hbuf = x, hbf = bf16(x), sq = rowwise ||x||^2 ----------------
__global__ void k_init(const float* __restrict__ x, float* __restrict__ hbuf,
                       bf16* __restrict__ hbf, float* __restrict__ sq) {
  int lane = threadIdx.x & 31, wid = threadIdx.x >> 5;
  size_t row = (size_t)blockIdx.x * 8 + wid;
  size_t base = row * CD;
  float ss = 0.f;
#pragma unroll
  for (int i = 0; i < 4; i++) {
    float v = x[base + lane + 32 * i];
    hbuf[base + lane + 32 * i] = v;
    hbf[base + lane + 32 * i] = (bf16)v;
    ss += v * v;
  }
  ss = wsum(ss);
  if (lane == 0) sq[row] = ss;
}

// ---------------- pairwise distances via WMMA Gram matrix ----------------
__global__ void k_gram(const bf16* __restrict__ hbf, const float* __restrict__ sq,
                       float* __restrict__ dis) {
  int lane = threadIdx.x;
  int nb = blockIdx.x * 32, mb = blockIdx.y * 32, b = blockIdx.z;
  const bf16* A = hbf + ((size_t)b * CN + nb) * CD;
  const bf16* Bp = hbf + ((size_t)b * CN + mb) * CD;
  v8f z = {0, 0, 0, 0, 0, 0, 0, 0};
  v8f c00 = z, c01 = z, c10 = z, c11 = z;
  for (int kk = 0; kk < CD; kk += 32) {
    v16bf a0 = loadA16(A, CD, lane, kk);
    v16bf a1 = loadA16(A + 16 * CD, CD, lane, kk);
    v16bf b0 = loadB16(Bp, CD, lane, kk);
    v16bf b1 = loadB16(Bp + 16 * CD, CD, lane, kk);
    c00 = wmma_bf16(a0, b0, c00);
    c01 = wmma_bf16(a0, b1, c01);
    c10 = wmma_bf16(a1, b0, c10);
    c11 = wmma_bf16(a1, b1, c11);
  }
  int col = lane & 15, p = lane >> 4;
  size_t bN = (size_t)b * CN;
  float sm0 = sq[bN + mb + col];
  float sm1 = sq[bN + mb + 16 + col];
#pragma unroll
  for (int r = 0; r < 8; r++) {
    int n0 = nb + r + 8 * p, n1 = n0 + 16;
    float sn0 = sq[bN + n0], sn1 = sq[bN + n1];
    dis[(bN + n0) * CN + mb + col]      = sn0 + sm0 - 2.f * c00[r];
    dis[(bN + n0) * CN + mb + 16 + col] = sn0 + sm1 - 2.f * c01[r];
    dis[(bN + n1) * CN + mb + col]      = sn1 + sm0 - 2.f * c10[r];
    dis[(bN + n1) * CN + mb + 16 + col] = sn1 + sm1 - 2.f * c11[r];
  }
}

// ---------------- top-K smallest per row -> neighbor bitmask ----------------
__global__ void k_topk(const float* __restrict__ dis, unsigned* __restrict__ kmask) {
  __shared__ float sd[8 * CN];
  volatile float* vs = sd;
  int lane = threadIdx.x & 31, wid = threadIdx.x >> 5;
  size_t row = (size_t)blockIdx.x * 8 + wid;
  const float* dr = dis + row * CN;
  int base = wid * CN;
#pragma unroll 4
  for (int j = 0; j < 32; j++) vs[base + j * 32 + lane] = dr[j * 32 + lane];
  __syncthreads();
  unsigned msk = 0;
  for (int t = 0; t < CK; t++) {
    float v = 3.0e38f;
    int mi = 0;
    for (int j = 0; j < 32; j++) {
      float d = vs[base + j * 32 + lane];
      int idx = j * 32 + lane;
      if (d < v) { v = d; mi = idx; }
    }
#pragma unroll
    for (int off = 16; off; off >>= 1) {
      float ov = __shfl_xor(v, off, 32);
      int oi = __shfl_xor(mi, off, 32);
      if (ov < v || (ov == v && oi < mi)) { v = ov; mi = oi; }
    }
    if (lane == 0) vs[base + mi] = 3.0e38f;
    if (lane == (mi >> 5)) msk |= (1u << (mi & 31));
    __syncthreads();
  }
  kmask[row * 32 + lane] = msk;
}

// ---------------- QKV projections (Q pre-scaled by 0.125*log2e; V stored transposed) ----------------
__global__ void k_qkv(const bf16* __restrict__ hbf, const bf16* __restrict__ wl,
                      bf16* __restrict__ Qb, bf16* __restrict__ Kb, bf16* __restrict__ Vt) {
  int lane = threadIdx.x;
  int nt = blockIdx.x, y = blockIdx.y, b = blockIdx.z;
  int h = y & 7, t = y >> 3;
  const bf16* W = wl + (size_t)t * (512 * CD) + (size_t)(h * 64) * CD;
  const bf16* A = hbf + ((size_t)b * CN + nt * 16) * CD;
  v8f z = {0, 0, 0, 0, 0, 0, 0, 0};
  v8f c[4] = {z, z, z, z};
  for (int kk = 0; kk < CD; kk += 32) {
    v16bf a = loadA16(A, CD, lane, kk);
#pragma unroll
    for (int j = 0; j < 4; j++) {
      v16bf bb = loadB16(W + (size_t)(16 * j) * CD, CD, lane, kk);
      c[j] = wmma_bf16(a, bb, c[j]);
    }
  }
  int col = lane & 15, p = lane >> 4;
  size_t bh = (size_t)b * CH + h;
  if (t == 2) {  // V transposed: Vt[b,h,dv,n]
#pragma unroll
    for (int j = 0; j < 4; j++) {
      v8bf pk;
#pragma unroll
      for (int r = 0; r < 8; r++) pk[r] = (bf16)c[j][r];
      *(v8bf*)(Vt + (bh * 64 + 16 * j + col) * CN + (size_t)nt * 16 + 8 * p) = pk;
    }
  } else {
    bf16* out = (t == 0) ? Qb : Kb;
    float scl = (t == 0) ? 0.18033688f : 1.0f;  // (1/8)*log2(e) folded into Q
#pragma unroll
    for (int j = 0; j < 4; j++)
#pragma unroll
      for (int r = 0; r < 8; r++) {
        int n = nt * 16 + r + 8 * p;
        out[(bh * CN + n) * 64 + 16 * j + col] = (bf16)(c[j][r] * scl);
      }
  }
}

// ---------------- flash attention with kNN mask ----------------
// 4 waves/block = 4 q-tiles; K/V key chunks staged in LDS, double-buffered via
// the Tensor Data Mover (tensor_load_to_lds + s_wait_tensorcnt) when available.
__global__ void k_attn(const bf16* __restrict__ Qb, const bf16* __restrict__ Kb,
                       const bf16* __restrict__ Vt, const unsigned* __restrict__ kmask,
                       bf16* __restrict__ obf) {
  __shared__ __align__(16) bf16 sk[2][32 * 64];  // keys chunk: [32 keys][64]
  __shared__ __align__(16) bf16 sv[2][64 * 32];  // V^T chunk:  [64 dv][32 keys]
  int tid = threadIdx.x;
  int lane = tid & 31, wid = tid >> 5;
  int qt = blockIdx.x * 4 + wid, h = blockIdx.y, b = blockIdx.z;
  int col = lane & 15, p = lane >> 4;
  int q = qt * 16 + col;
  size_t bh = (size_t)b * CH + h;
  const bf16* Kbase = Kb + bh * CN * 64;
  const bf16* Vbase = Vt + bh * 64 * CN;
  // Q^T B-operand, reused for all key chunks
  v16bf bq0 = loadB16(Qb + (bh * CN + (size_t)qt * 16) * 64, 64, lane, 0);
  v16bf bq1 = loadB16(Qb + (bh * CN + (size_t)qt * 16) * 64, 64, lane, 32);
  const unsigned* mrow = kmask + ((size_t)b * CN + q) * 32;
  v8f z = {0, 0, 0, 0, 0, 0, 0, 0};
  v8f ot[4] = {z, z, z, z};
  float m = -1e30f, lsum = 0.f;
  constexpr int NCH = CN / 32;
#if USE_TDM
  if (tid < 32) {  // wave 0 drives the TDM
    tdm_load_2d(lds_off_of(sk[0]), Kbase, 64, 32, 64, CN, 64ull);
    tdm_load_2d(lds_off_of(sv[0]), Vbase, 32, 64, CN, 64, (unsigned long long)CN);
  }
#endif
  for (int ic = 0; ic < NCH; ic++) {
#if USE_TDM
    int buf = ic & 1;
    if (tid < 32) {
      if (ic + 1 < NCH) {
        tdm_load_2d(lds_off_of(sk[buf ^ 1]), Kbase + (size_t)(ic + 1) * 32 * 64, 64, 32, 64, CN, 64ull);
        tdm_load_2d(lds_off_of(sv[buf ^ 1]), Vbase + (size_t)(ic + 1) * 32, 32, 64, CN, 64,
                    (unsigned long long)CN);
        __builtin_amdgcn_s_wait_tensorcnt(2);  // chunk ic landed
      } else {
        __builtin_amdgcn_s_wait_tensorcnt(0);
      }
    }
    __syncthreads();
#else
    int buf = 0;
    __syncthreads();
    {  // cooperative staging
      const bf16* gk = Kbase + (size_t)ic * 32 * 64;
      for (int i = tid; i < 32 * 64 / 8; i += 128) ((v8bf*)sk[0])[i] = ((const v8bf*)gk)[i];
      for (int i = tid; i < 64 * 32 / 8; i += 128) {
        int row = i >> 2, cp = i & 3;
        *(v8bf*)(sv[0] + row * 32 + cp * 8) =
            *(const v8bf*)(Vbase + (size_t)row * CN + (size_t)ic * 32 + cp * 8);
      }
    }
    __syncthreads();
#endif
    const bf16* kb = sk[buf];
    const bf16* vb = sv[buf];
    v8f s0 = z, s1 = z;
    {
      v16bf a = loadA16(kb, 64, lane, 0);
      s0 = wmma_bf16(a, bq0, s0);
      a = loadA16(kb, 64, lane, 32);
      s0 = wmma_bf16(a, bq1, s0);
      a = loadA16(kb + 16 * 64, 64, lane, 0);
      s1 = wmma_bf16(a, bq0, s1);
      a = loadA16(kb + 16 * 64, 64, lane, 32);
      s1 = wmma_bf16(a, bq1, s1);
    }
    unsigned mw = mrow[ic];
    float e0[8], e1[8];
    float mx = -1e30f;
#pragma unroll
    for (int r = 0; r < 8; r++) {
      int b0 = r + 8 * p, b1 = 16 + r + 8 * p;
      float v0 = ((mw >> b0) & 1u) ? s0[r] : -1e30f;
      float v1 = ((mw >> b1) & 1u) ? s1[r] : -1e30f;
      e0[r] = v0; e1[r] = v1;
      mx = fmaxf(mx, fmaxf(v0, v1));
    }
    mx = fmaxf(mx, __shfl_xor(mx, 16, 32));
    float mnew = fmaxf(m, mx);
    float alpha = exp2f(m - mnew);
    float s = 0.f;
#pragma unroll
    for (int r = 0; r < 8; r++) {
      e0[r] = exp2f(e0[r] - mnew);
      e1[r] = exp2f(e1[r] - mnew);
      s += e0[r] + e1[r];
    }
    s += __shfl_xor(s, 16, 32);
    lsum = lsum * alpha + s;
    m = mnew;
#pragma unroll
    for (int j = 0; j < 4; j++)
#pragma unroll
      for (int r = 0; r < 8; r++) ot[j][r] *= alpha;
    // pack P^T as B-operand (32x16): lanes 0-15 hold k=0..15, lanes 16-31 hold k=16..31
    v16bf pb;
#pragma unroll
    for (int r = 0; r < 8; r++) {
      float o0 = __shfl_xor(e0[r], 16, 32);
      float o1 = __shfl_xor(e1[r], 16, 32);
      pb[r]     = (bf16)(p ? o1 : e0[r]);
      pb[8 + r] = (bf16)(p ? e1[r] : o0);
    }
#pragma unroll
    for (int j = 0; j < 4; j++) {
      v16bf av = loadA16(vb + (size_t)(16 * j) * 32, 32, lane, 0);
      ot[j] = wmma_bf16(av, pb, ot[j]);
    }
    __syncthreads();
  }
  float rinv = 1.0f / lsum;
#pragma unroll
  for (int j = 0; j < 4; j++) {
    v8bf pk;
#pragma unroll
    for (int r = 0; r < 8; r++) pk[r] = (bf16)(ot[j][r] * rinv);
    *(v8bf*)(obf + ((size_t)b * CN + q) * 512 + h * 64 + 16 * j + 8 * p) = pk;
  }
}

// ---------------- generic WMMA GEMMs ----------------
__global__ void k_wo(const bf16* __restrict__ obf, const bf16* __restrict__ WOt,
                     float* __restrict__ yraw) {
  int lane = threadIdx.x;
  size_t row0 = (size_t)blockIdx.x * 16;
  int ct = blockIdx.y;
  const bf16* A = obf + row0 * 512;
  v8f z = {0, 0, 0, 0, 0, 0, 0, 0};
  v8f c[2] = {z, z};
  for (int kk = 0; kk < 512; kk += 32) {
    v16bf a = loadA16(A, 512, lane, kk);
#pragma unroll
    for (int j = 0; j < 2; j++) {
      v16bf bb = loadB16(WOt + (size_t)(ct * 32 + 16 * j) * 512, 512, lane, kk);
      c[j] = wmma_bf16(a, bb, c[j]);
    }
  }
  int col = lane & 15, p = lane >> 4;
#pragma unroll
  for (int j = 0; j < 2; j++)
#pragma unroll
    for (int r = 0; r < 8; r++)
      yraw[(row0 + r + 8 * p) * CD + ct * 32 + 16 * j + col] = c[j][r];
}

__global__ void k_ff1(const bf16* __restrict__ ybf, const bf16* __restrict__ F1t,
                      bf16* __restrict__ tbuf) {
  int lane = threadIdx.x;
  size_t row0 = (size_t)blockIdx.x * 16;
  int ct = blockIdx.y;
  const bf16* A = ybf + row0 * CD;
  v8f z = {0, 0, 0, 0, 0, 0, 0, 0};
  v8f c[4] = {z, z, z, z};
  for (int kk = 0; kk < CD; kk += 32) {
    v16bf a = loadA16(A, CD, lane, kk);
#pragma unroll
    for (int j = 0; j < 4; j++) {
      v16bf bb = loadB16(F1t + (size_t)(ct * 64 + 16 * j) * CD, CD, lane, kk);
      c[j] = wmma_bf16(a, bb, c[j]);
    }
  }
  int col = lane & 15, p = lane >> 4;
#pragma unroll
  for (int j = 0; j < 4; j++)
#pragma unroll
    for (int r = 0; r < 8; r++) {
      float v = fmaxf(c[j][r], 0.f);  // relu
      tbuf[(row0 + r + 8 * p) * CMID + ct * 64 + 16 * j + col] = (bf16)v;
    }
}

__global__ void k_ff2(const bf16* __restrict__ tbuf, const bf16* __restrict__ F2t,
                      float* __restrict__ zraw) {
  int lane = threadIdx.x;
  size_t row0 = (size_t)blockIdx.x * 16;
  int ct = blockIdx.y;
  const bf16* A = tbuf + row0 * CMID;
  v8f z = {0, 0, 0, 0, 0, 0, 0, 0};
  v8f c[2] = {z, z};
  for (int kk = 0; kk < CMID; kk += 32) {
    if (kk + 64 < CMID)
      __builtin_prefetch((const void*)(A + (size_t)(lane & 15) * CMID + kk + 64), 0, 1);
    v16bf a = loadA16(A, CMID, lane, kk);
#pragma unroll
    for (int j = 0; j < 2; j++) {
      v16bf bb = loadB16(F2t + (size_t)(ct * 32 + 16 * j) * CMID, CMID, lane, kk);
      c[j] = wmma_bf16(a, bb, c[j]);
    }
  }
  int col = lane & 15, p = lane >> 4;
#pragma unroll
  for (int j = 0; j < 2; j++)
#pragma unroll
    for (int r = 0; r < 8; r++)
      zraw[(row0 + r + 8 * p) * CD + ct * 32 + 16 * j + col] = c[j][r];
}

// ---------------- residual + LayerNorm (+ bf16 copy, + optional ||.||^2) ----------------
__global__ void k_addln(const float* __restrict__ a, const float* __restrict__ bsrc,
                        float* __restrict__ yout, bf16* __restrict__ ybout,
                        float* __restrict__ sqout) {
  int lane = threadIdx.x & 31, wid = threadIdx.x >> 5;
  size_t row = (size_t)blockIdx.x * 8 + wid;
  size_t base = row * CD;
  float u[4];
  float s = 0.f;
#pragma unroll
  for (int i = 0; i < 4; i++) {
    u[i] = a[base + lane + 32 * i] + bsrc[base + lane + 32 * i];
    s += u[i];
  }
  float mean = wsum(s) * (1.f / 128.f);
  float vv = 0.f;
#pragma unroll
  for (int i = 0; i < 4; i++) { u[i] -= mean; vv += u[i] * u[i]; }
  vv = wsum(vv) * (1.f / 128.f);
  float rs = rsqrtf(vv + 1e-5f);
  float ss = 0.f;
#pragma unroll
  for (int i = 0; i < 4; i++) {
    float y = u[i] * rs;
    yout[base + lane + 32 * i] = y;
    ybout[base + lane + 32 * i] = (bf16)y;
    ss += y * y;
  }
  if (sqout) {
    ss = wsum(ss);
    if (lane == 0) sqout[row] = ss;
  }
}

extern "C" void kernel_launch(void* const* d_in, const int* in_sizes, int n_in,
                              void* d_out, int out_size, void* d_ws, size_t ws_size,
                              hipStream_t stream) {
  (void)in_sizes; (void)n_in; (void)out_size; (void)ws_size;
  const float* x  = (const float*)d_in[0];
  const float* WQ = (const float*)d_in[1];
  const float* WK = (const float*)d_in[2];
  const float* WV = (const float*)d_in[3];
  const float* WO = (const float*)d_in[4];
  const float* F1 = (const float*)d_in[5];
  const float* F2 = (const float*)d_in[6];

  char* p = (char*)d_ws;
  auto carve = [&](size_t bytes) -> char* {
    char* r = p;
    p += (bytes + 255) & ~size_t(255);
    return r;
  };
  float* hbuf = (float*)carve((size_t)CB * CN * CD * 4);
  float* ybuf = (float*)carve((size_t)CB * CN * CD * 4);
  float* yraw = (float*)carve((size_t)CB * CN * CD * 4);
  float* zraw = (float*)carve((size_t)CB * CN * CD * 4);
  float* sq   = (float*)carve((size_t)CB * CN * 4);
  bf16* hbf = (bf16*)carve((size_t)CB * CN * CD * 2);
  bf16* ybf = (bf16*)carve((size_t)CB * CN * CD * 2);
  bf16* Qb  = (bf16*)carve((size_t)CB * CH * CN * 64 * 2);
  bf16* Kb  = (bf16*)carve((size_t)CB * CH * CN * 64 * 2);
  bf16* Vt  = (bf16*)carve((size_t)CB * CH * CN * 64 * 2);
  bf16* obf = (bf16*)carve((size_t)CB * CN * 512 * 2);
  unsigned* kmask = (unsigned*)carve((size_t)CB * CN * 32 * 4);
  bf16* wbf = (bf16*)carve((size_t)CL * W_LAYER * 2);
  char* big = carve((size_t)CB * CN * CN * 4);  // dis (f32) and tbuf (bf16) share this region
  float* dis = (float*)big;
  bf16* tbuf = (bf16*)big;

  // weight convert+transpose (bf16)
  for (int l = 0; l < CL; l++) {
    bf16* wl = wbf + (size_t)l * W_LAYER;
    k_transcvt<<<(CD * 512 + 255) / 256, 256, 0, stream>>>(WQ + (size_t)l * CD * 512, wl + W_QT, CD, 512);
    k_transcvt<<<(CD * 512 + 255) / 256, 256, 0, stream>>>(WK + (size_t)l * CD * 512, wl + W_KT, CD, 512);
    k_transcvt<<<(CD * 512 + 255) / 256, 256, 0, stream>>>(WV + (size_t)l * CD * 512, wl + W_VT, CD, 512);
    k_transcvt<<<(512 * CD + 255) / 256, 256, 0, stream>>>(WO + (size_t)l * 512 * CD, wl + W_OT, 512, CD);
    k_transcvt<<<(CD * CMID + 255) / 256, 256, 0, stream>>>(F1 + (size_t)l * CD * CMID, wl + W_F1, CD, CMID);
    k_transcvt<<<(CMID * CD + 255) / 256, 256, 0, stream>>>(F2 + (size_t)l * CMID * CD, wl + W_F2, CMID, CD);
  }
  k_init<<<CB * CN / 8, 256, 0, stream>>>(x, hbuf, hbf, sq);

  for (int l = 0; l < CL; l++) {
    bf16* wl = wbf + (size_t)l * W_LAYER;
    k_gram<<<dim3(CN / 32, CN / 32, CB), 32, 0, stream>>>(hbf, sq, dis);
    k_topk<<<CB * CN / 8, 256, 0, stream>>>(dis, kmask);
    k_qkv<<<dim3(CN / 16, CH * 3, CB), 32, 0, stream>>>(hbf, wl, Qb, Kb, Vt);
    k_attn<<<dim3(CN / 64, CH, CB), 128, 0, stream>>>(Qb, Kb, Vt, kmask, obf);
    k_wo<<<dim3(CB * CN / 16, CD / 32), 32, 0, stream>>>(obf, wl + W_OT, yraw);
    k_addln<<<CB * CN / 8, 256, 0, stream>>>(hbuf, yraw, ybuf, ybf, (float*)nullptr);
    k_ff1<<<dim3(CB * CN / 16, CMID / 64), 32, 0, stream>>>(ybf, wl + W_F1, tbuf);
    k_ff2<<<dim3(CB * CN / 16, CD / 32), 32, 0, stream>>>(tbuf, wl + W_F2, zraw);
    k_addln<<<CB * CN / 8, 256, 0, stream>>>(ybuf, zraw, hbuf, hbf, sq);
  }
  (void)hipMemcpyAsync(d_out, hbuf, (size_t)CB * CN * CD * 4, hipMemcpyDeviceToDevice, stream);
}
